// Transformer_13417477833533
// MI455X (gfx1250) — matrix operand order repaired and checked by reading, not compiled, not run on hardware
//
#include <hip/hip_runtime.h>
#include <stdint.h>

typedef _Float16 v16h __attribute__((ext_vector_type(16)));
typedef _Float16 v8h  __attribute__((ext_vector_type(8)));
typedef float    v8f  __attribute__((ext_vector_type(8)));
typedef int      v4i  __attribute__((ext_vector_type(4)));
typedef __attribute__((address_space(1))) v4i gv4i;   // global
typedef __attribute__((address_space(3))) v4i lv4i;   // LDS

#define DIM    128
#define M_NB   16
#define DHEAD  32
#define TILE   16
#define KVPITCH 264   // 256 + 8 halves pad (4-dword bank shift per row)
#define ATTN_SCALE 0.17677669529663687f   // 32^-0.5
#define LN_EPS 1e-5f

// ---- LDS layout (bytes) ----
constexpr unsigned OFF_WKVT = 0;                   // 256x128 f16          = 65536
constexpr unsigned OFF_LNG  = OFF_WKVT + 65536;    // 128 f32
constexpr unsigned OFF_LNB  = OFF_LNG  + 512;
constexpr unsigned OFF_BO   = OFF_LNB  + 512;
constexpr unsigned OFF_XNA  = OFF_BO   + 512;      // 16x128 f16           = 4096
constexpr unsigned OFF_QBUF = OFF_XNA  + 4096;     // 16x128 f32           = 8192
constexpr unsigned OFF_YBUF = OFF_QBUF + 8192;     // 8 x 16x128 f16       = 32768
constexpr unsigned OFF_KV   = OFF_YBUF + 32768;    // 8 x 16x264 f16       = 67584
constexpr unsigned OFF_ABUF = OFF_KV   + 67584;    // 8 x 64 f32           = 2048
constexpr unsigned OFF_OBUF = OFF_ABUF + 2048;     // 16x128 f16           = 4096
constexpr unsigned OFF_YST  = OFF_OBUF + 4096;     // 8 x 2 x 16x128 f32   = 131072
constexpr unsigned SMEM_BYTES = OFF_YST + 131072;  // = 316928 (<= 320KB WGP LDS)

#if defined(__has_builtin)
# if __has_builtin(__builtin_amdgcn_global_load_async_to_lds_b128)
#  define USE_ASYNC_BUILTIN 1
# endif
# if __has_builtin(__builtin_amdgcn_s_wait_asynccnt)
#  define USE_WAITASYNC_BUILTIN 1
# endif
#endif

__device__ __forceinline__ void wait_async16() {
#ifdef USE_WAITASYNC_BUILTIN
  __builtin_amdgcn_s_wait_asynccnt(16);
#else
  asm volatile("s_wait_asynccnt 0x10" ::: "memory");
#endif
}
__device__ __forceinline__ void wait_async0() {
#ifdef USE_WAITASYNC_BUILTIN
  __builtin_amdgcn_s_wait_asynccnt(0);
#else
  asm volatile("s_wait_asynccnt 0x0" ::: "memory");
#endif
}

// One 128-float row, global -> LDS, 16B per lane (ASYNCcnt-tracked).
__device__ __forceinline__ void async_y_row(const float* gsrc, char* smem,
                                            unsigned lds_byte_off, int lane) {
  const float* gp = gsrc + lane * 4;
  char* lp = smem + lds_byte_off + (unsigned)lane * 16u;
#ifdef USE_ASYNC_BUILTIN
  __builtin_amdgcn_global_load_async_to_lds_b128((gv4i*)gp, (lv4i*)lp, 0, 0);
#else
  const unsigned lo = lds_byte_off + (unsigned)lane * 16u;
  asm volatile("global_load_async_to_lds_b128 %0, %1, off"
               :: "v"(lo), "v"(gp) : "memory");
#endif
}

union AF { v16h v; v8h h[2]; };

// A-fragment: 16x32 f16, row-major source (row stride = DIM halves).
// Contiguous k order (not AMD's table order): lanes 0-15 -> K base 0, lanes 16-31 -> K base 16;
// halves 0..15 -> K base+0..15.
__device__ __forceinline__ v16h load_a_frag(const _Float16* A, int lane, int ks) {
  const int row = lane & 15;
  const int g   = (lane & 16) ? 16 : 0;
  const _Float16* p = A + row * DIM + ks * 32 + g;
  AF f;
  f.h[0] = *(const v8h*)(p);
  f.h[1] = *(const v8h*)(p + 8);
  return f.v;
}

// B-fragment: 32x16 f16 from [N][K]-major weights (row stride = DIM halves).
// Contiguous k order, same as load_a_frag: lanes 0-15 hold K 0-15, lanes 16-31 hold K 16-31; N = lane%16.
__device__ __forceinline__ v16h load_b_frag(const _Float16* WT, int lane, int ntile, int ks) {
  const int n    = ntile * 16 + (lane & 15);
  const int koff = ks * 32 + ((lane & 16) ? 16 : 0);
  return *(const v16h*)(WT + n * DIM + koff);
}

__device__ __forceinline__ v8f wmma_f16(v16h a, v16h b, v8f c) {
  return __builtin_amdgcn_wmma_f32_16x16x32_f16(
      false, a, false, b, (short)0, c, false, false);
}

// Wave32 LayerNorm, two rows per call: lanes 0-15 own one row, lanes 16-31 the
// other; each lane covers 8 columns. Reductions need only xor masks 8/4/2/1
// (stay inside each 16-lane group). Output: one ds_store_b128 per lane.
// src/dst are the ROW base pointers already selected for this lane's half.
__device__ __forceinline__ void layernorm_2rows(const float* __restrict__ src,
                                                _Float16* dst,
                                                const float* g, const float* b,
                                                int lane) {
  const int c0 = (lane & 15) * 8;
  const float4 va = *(const float4*)(src + c0);
  const float4 vb = *(const float4*)(src + c0 + 4);
  float s = (va.x + va.y) + (va.z + va.w) + (vb.x + vb.y) + (vb.z + vb.w);
  #pragma unroll
  for (int m = 8; m >= 1; m >>= 1) s += __shfl_xor(s, m, 32);
  const float mu = s * (1.0f / 128.0f);
  const float d0 = va.x - mu, d1 = va.y - mu, d2 = va.z - mu, d3 = va.w - mu;
  const float d4 = vb.x - mu, d5 = vb.y - mu, d6 = vb.z - mu, d7 = vb.w - mu;
  float q = (d0 * d0 + d1 * d1) + (d2 * d2 + d3 * d3) +
            (d4 * d4 + d5 * d5) + (d6 * d6 + d7 * d7);
  #pragma unroll
  for (int m = 8; m >= 1; m >>= 1) q += __shfl_xor(q, m, 32);
  const float rstd = rsqrtf(q * (1.0f / 128.0f) + LN_EPS);
  v8h o;
  o[0] = (_Float16)(d0 * rstd * g[c0 + 0] + b[c0 + 0]);
  o[1] = (_Float16)(d1 * rstd * g[c0 + 1] + b[c0 + 1]);
  o[2] = (_Float16)(d2 * rstd * g[c0 + 2] + b[c0 + 2]);
  o[3] = (_Float16)(d3 * rstd * g[c0 + 3] + b[c0 + 3]);
  o[4] = (_Float16)(d4 * rstd * g[c0 + 4] + b[c0 + 4]);
  o[5] = (_Float16)(d5 * rstd * g[c0 + 5] + b[c0 + 5]);
  o[6] = (_Float16)(d6 * rstd * g[c0 + 6] + b[c0 + 6]);
  o[7] = (_Float16)(d7 * rstd * g[c0 + 7] + b[c0 + 7]);
  *(v8h*)(dst + c0) = o;
}

// Once per launch: Wq/Wo f32 [K][N] -> f16 [N][K] in global scratch.
__global__ __launch_bounds__(256) void prep_weights(const float* __restrict__ Wq,
                                                    const float* __restrict__ Wo,
                                                    _Float16* __restrict__ wsq,
                                                    _Float16* __restrict__ wso) {
  const int i = blockIdx.x * 256 + threadIdx.x;
  if (i < 128 * 128) {
    wsq[(i & 127) * 128 + (i >> 7)] = (_Float16)Wq[i];
    wso[(i & 127) * 128 + (i >> 7)] = (_Float16)Wo[i];
  }
}

__global__ __launch_bounds__(256, 1)
void ca_fused_kernel(const float* __restrict__ x,  const float* __restrict__ y,
                     const float* __restrict__ ln_g, const float* __restrict__ ln_b,
                     const _Float16* __restrict__ wsq, const float* __restrict__ Wkv,
                     const _Float16* __restrict__ wso, const float* __restrict__ bo,
                     float* __restrict__ out, int ntiles) {
  extern __shared__ char smem[];
  _Float16* WkvT  = (_Float16*)(smem + OFF_WKVT);
  float*    lng   = (float*)(smem + OFF_LNG);
  float*    lnb   = (float*)(smem + OFF_LNB);
  float*    bos   = (float*)(smem + OFF_BO);
  _Float16* xnA   = (_Float16*)(smem + OFF_XNA);
  float*    qbuf  = (float*)(smem + OFF_QBUF);
  _Float16* ybuf  = (_Float16*)(smem + OFF_YBUF);
  _Float16* kvbuf = (_Float16*)(smem + OFF_KV);
  float*    abuf  = (float*)(smem + OFF_ABUF);
  _Float16* obuf  = (_Float16*)(smem + OFF_OBUF);

  const int tid  = threadIdx.x;
  const int w    = tid >> 5;     // wave id 0..7
  const int lane = tid & 31;
  const int lhalf = lane >> 4;   // 0 or 1: which row of a row-pair this lane owns

  // ---- One-time: Wkv -> LDS f16 transposed [N][K]; LN/bias vectors ----
  for (int i = tid; i < 128 * 256; i += 256)
    WkvT[(i & 255) * 128 + (i >> 8)] = (_Float16)Wkv[i];
  if (tid < 128) { lng[tid] = ln_g[tid]; lnb[tid] = ln_b[tid]; bos[tid] = bo[tid]; }

  // ---- Loop-invariant Wq/Wo B-fragments for this wave's n-tile -> registers ----
  v16h bwq[4], bwo[4];
  #pragma unroll
  for (int ks = 0; ks < 4; ++ks) {
    bwq[ks] = load_b_frag(wsq, lane, w, ks);
    bwo[ks] = load_b_frag(wso, lane, w, ks);
  }
  __syncthreads();

  _Float16* ybufw = ybuf  + w * (16 * DIM);
  _Float16* kvw   = kvbuf + w * (16 * KVPITCH);
  float*    abufw = abuf  + w * 64;
  const unsigned st0 = OFF_YST + (unsigned)w * 16384u;   // stage for center 2w
  const unsigned st1 = st0 + 8192u;                      // stage for center 2w+1
  const float* stg0 = (const float*)(smem + st0);
  const float* stg1 = (const float*)(smem + st1);
  const int rb = (lane & 16) ? 8 : 0;   // D-matrix row base for this lane half
  const int cl = lane & 15;             // D-matrix column within n-tile

  // ---- Prime the async y pipeline for the first tile (stage0 first!) ----
  {
    const int gc0 = blockIdx.x * TILE + 2 * w;
    for (int j = 0; j < 16; ++j)
      async_y_row(y + ((size_t)gc0 * M_NB + j) * DIM, smem, st0 + j * 512u, lane);
    for (int j = 0; j < 16; ++j)
      async_y_row(y + ((size_t)(gc0 + 1) * M_NB + j) * DIM, smem, st1 + j * 512u, lane);
  }

  for (int tile = blockIdx.x; tile < ntiles; tile += gridDim.x) {
    const int c0  = tile * TILE;
    const bool pf = (tile + (int)gridDim.x) < ntiles;
    const int c0n = (tile + (int)gridDim.x) * TILE;   // next grid-stride tile

    // ---- Phase 1: LayerNorm(x) -> xnA (wave w owns rows 2w, 2w+1) ----
    {
      const int r = 2 * w + lhalf;
      layernorm_2rows(x + (size_t)(c0 + r) * DIM, xnA + r * DIM, lng, lnb, lane);
      if (pf) __builtin_prefetch(x + (size_t)(c0n + r) * DIM + (lane & 15) * 8, 0, 0);
    }
    __syncthreads();

    // ---- Phase 2: q = (xn @ Wq) * scale ; wave w owns n-tile w ----
    {
      v16h a0 = load_a_frag(xnA, lane, 0);
      v16h a1 = load_a_frag(xnA, lane, 1);
      v16h a2 = load_a_frag(xnA, lane, 2);
      v16h a3 = load_a_frag(xnA, lane, 3);
      v8f acc = {};
      acc = wmma_f16(a0, bwq[0], acc);
      acc = wmma_f16(a1, bwq[1], acc);
      acc = wmma_f16(a2, bwq[2], acc);
      acc = wmma_f16(a3, bwq[3], acc);
      const int col = w * 16 + cl;
      #pragma unroll
      for (int r = 0; r < 8; ++r)
        qbuf[(r + rb) * DIM + col] = acc[r] * ATTN_SCALE;
    }
    __syncthreads();

    // ---- Phase 3: per-wave centers, double-buffered async y staging ----
    auto process_center = [&](int lc, const float* stg, unsigned stoff) {
      // LN neighbors from LDS stage -> f16 WMMA A-tile (2 rows per pass)
      #pragma unroll
      for (int jp = 0; jp < 8; ++jp) {
        const int j = 2 * jp + lhalf;
        layernorm_2rows(stg + j * DIM, ybufw + j * DIM, lng, lnb, lane);
      }
      // stage consumed: refill it with next tile's block (overlaps kv GEMM)
      if (pf) {
        const float* gnext = y + ((size_t)(c0n + lc) * M_NB) * DIM;
        for (int j = 0; j < 16; ++j)
          async_y_row(gnext + j * DIM, smem, stoff + j * 512u, lane);
      }
      // kv = yn @ Wkv : 16 n-tiles x 4 k-steps = 64 WMMAs (fully unrolled so
      // the scheduler overlaps B-fragment ds_loads with in-flight WMMAs)
      {
        v16h a0 = load_a_frag(ybufw, lane, 0);
        v16h a1 = load_a_frag(ybufw, lane, 1);
        v16h a2 = load_a_frag(ybufw, lane, 2);
        v16h a3 = load_a_frag(ybufw, lane, 3);
        #pragma unroll
        for (int nt = 0; nt < 16; ++nt) {
          v8f acc = {};
          acc = wmma_f16(a0, load_b_frag(WkvT, lane, nt, 0), acc);
          acc = wmma_f16(a1, load_b_frag(WkvT, lane, nt, 1), acc);
          acc = wmma_f16(a2, load_b_frag(WkvT, lane, nt, 2), acc);
          acc = wmma_f16(a3, load_b_frag(WkvT, lane, nt, 3), acc);
          #pragma unroll
          for (int r = 0; r < 8; ++r)
            kvw[(r + rb) * KVPITCH + nt * 16 + cl] = (_Float16)acc[r];
        }
      }
      // dots + softmax: lane -> (h = lane/16 + 2*hh, j = lane%16)
      const int j = lane & 15;
      #pragma unroll
      for (int hh = 0; hh < 2; ++hh) {
        const int h = lhalf + 2 * hh;
        const float*    qp = qbuf + lc * DIM + h * DHEAD;   // scale folded in
        const _Float16* kp = kvw + j * KVPITCH + h * DHEAD;
        float dot = 0.0f;
        #pragma unroll
        for (int d = 0; d < DHEAD; ++d) dot += qp[d] * (float)kp[d];
        float mx = dot;
        #pragma unroll
        for (int m = 8; m >= 1; m >>= 1) mx = fmaxf(mx, __shfl_xor(mx, m, 32));
        const float e = __expf(dot - mx);
        float sum = e;
        #pragma unroll
        for (int m = 8; m >= 1; m >>= 1) sum += __shfl_xor(sum, m, 32);
        abufw[h * 16 + j] = e / sum;
      }
      // out_inner[i] = sum_j attn[h=i/32][j] * v[j][i] -> obuf (f16)
      #pragma unroll
      for (int p = 0; p < 4; ++p) {
        const int i = p * 32 + lane;           // h == p for all lanes this pass
        float o = 0.0f;
        #pragma unroll
        for (int jj = 0; jj < 16; ++jj)
          o += abufw[p * 16 + jj] * (float)kvw[jj * KVPITCH + 128 + i];
        obuf[lc * DIM + i] = (_Float16)o;
      }
    };

    wait_async16();                       // stage0 (oldest 16 asyncs) landed
    process_center(2 * w, stg0, st0);
    if (pf) wait_async16(); else wait_async0();   // stage1 landed
    process_center(2 * w + 1, stg1, st1);
    __syncthreads();

    // ---- Phase 4: out = attn_out @ Wo + bo + x (residual) ----
    {
      v16h a0 = load_a_frag(obuf, lane, 0);
      v16h a1 = load_a_frag(obuf, lane, 1);
      v16h a2 = load_a_frag(obuf, lane, 2);
      v16h a3 = load_a_frag(obuf, lane, 3);
      v8f acc = {};
      acc = wmma_f16(a0, bwo[0], acc);
      acc = wmma_f16(a1, bwo[1], acc);
      acc = wmma_f16(a2, bwo[2], acc);
      acc = wmma_f16(a3, bwo[3], acc);
      const int col = w * 16 + cl;
      #pragma unroll
      for (int r = 0; r < 8; ++r) {
        const int gc = c0 + r + rb;
        out[(size_t)gc * DIM + col] = acc[r] + bos[col] + x[(size_t)gc * DIM + col];
      }
    }
    __syncthreads();
  }
}

extern "C" void kernel_launch(void* const* d_in, const int* in_sizes, int n_in,
                              void* d_out, int out_size, void* d_ws, size_t ws_size,
                              hipStream_t stream) {
  (void)n_in; (void)out_size; (void)ws_size;
  const float* x    = (const float*)d_in[0];
  const float* y    = (const float*)d_in[1];
  const float* ln_g = (const float*)d_in[2];
  const float* ln_b = (const float*)d_in[3];
  const float* Wq   = (const float*)d_in[4];
  const float* Wkv  = (const float*)d_in[5];
  const float* Wo   = (const float*)d_in[6];
  const float* bo   = (const float*)d_in[7];
  float* out = (float*)d_out;

  _Float16* wsq = (_Float16*)d_ws;                       // 128x128 f16 [N][K]
  _Float16* wso = (_Float16*)((char*)d_ws + 32768);      // 128x128 f16 [N][K]

  const int tokens = in_sizes[0] / DIM;   // b*n = 65536
  const int ntiles = tokens / TILE;       // 4096
  int grid = ntiles < 512 ? ntiles : 512; // grid-stride; amortizes weight staging

  prep_weights<<<dim3(64), dim3(256), 0, stream>>>(Wq, Wo, wsq, wso);
  ca_fused_kernel<<<dim3(grid), dim3(256), SMEM_BYTES, stream>>>(
      x, y, ln_g, ln_b, wsq, Wkv, wso, bo, out, ntiles);
}
